// DynamicGATLayer_47330539602431
// MI455X (gfx1250) — compile-verified, matrix operand order checked by reading, not run on hardware
//
#include <hip/hip_runtime.h>
#include <hip/hip_bf16.h>
#include <math.h>

// ---------------- problem constants (from reference) ----------------
#define NN      50000
#define EE      1600000
#define ETOT    (EE + NN)      // edges + self loops
#define IN_C    256
#define OUT_C   64
#define HEADS   2
#define HC      (HEADS * OUT_C)   // 128
#define G3      (3 * OUT_C)       // 192
#define NEG_SLOPE 0.2f

typedef __attribute__((ext_vector_type(16))) __bf16 v16bf;
typedef __attribute__((ext_vector_type(8)))  float  v8f;

// ---------------------------------------------------------------
// Build one 16-bit WMMA fragment (A-style or B-style: both are
// "16 bf16 values per lane, two contiguous 8-float runs" in our
// row-major/K-major storage):
//   elements 0..7  <- floats [kbase,      kbase+8)
//   elements 8..15 <- floats [kbase+16,   kbase+24)
// 4 x 32B-aligned float4 loads -> global_load_b128.
// ---------------------------------------------------------------
template<bool HAS_BIAS>
__device__ __forceinline__ v16bf load_frag(const float* __restrict__ base,
                                           const float* __restrict__ bias,
                                           int kbase)
{
    float4 f0 = *(const float4*)(base + kbase);
    float4 f1 = *(const float4*)(base + kbase + 4);
    float4 f2 = *(const float4*)(base + kbase + 16);
    float4 f3 = *(const float4*)(base + kbase + 20);
    if (HAS_BIAS) {
        const float4 b0 = *(const float4*)(bias + kbase);
        const float4 b1 = *(const float4*)(bias + kbase + 4);
        const float4 b2 = *(const float4*)(bias + kbase + 16);
        const float4 b3 = *(const float4*)(bias + kbase + 20);
        f0.x += b0.x; f0.y += b0.y; f0.z += b0.z; f0.w += b0.w;
        f1.x += b1.x; f1.y += b1.y; f1.z += b1.z; f1.w += b1.w;
        f2.x += b2.x; f2.y += b2.y; f2.z += b2.z; f2.w += b2.w;
        f3.x += b3.x; f3.y += b3.y; f3.z += b3.z; f3.w += b3.w;
    }
    v16bf r;
    r[0]  = (__bf16)f0.x; r[1]  = (__bf16)f0.y; r[2]  = (__bf16)f0.z; r[3]  = (__bf16)f0.w;
    r[4]  = (__bf16)f1.x; r[5]  = (__bf16)f1.y; r[6]  = (__bf16)f1.z; r[7]  = (__bf16)f1.w;
    r[8]  = (__bf16)f2.x; r[9]  = (__bf16)f2.y; r[10] = (__bf16)f2.z; r[11] = (__bf16)f2.w;
    r[12] = (__bf16)f3.x; r[13] = (__bf16)f3.y; r[14] = (__bf16)f3.z; r[15] = (__bf16)f3.w;
    return r;
}

// ---------------------------------------------------------------
// fp32-in / fp32-out GEMM via bf16 WMMA:
//   Out[M,Ncols] = (A[M,K] + a_bias[K]) @ W[Ncols,K]^T + out_bias[Ncols]
// One wave computes a 16x64 tile (4 accumulators sharing one A fragment:
// 4x less A traffic, 4 back-to-back WMMAs per fully-unrolled k-step).
// K: compile-time multiple of 32. Ncols: multiple of 64. M: multiple of 16.
// ---------------------------------------------------------------
template<int K, bool HAS_ABIAS>
__global__ void wmma_gemm_bf16(const float* __restrict__ A,
                               const float* __restrict__ W,
                               const float* __restrict__ a_bias,
                               const float* __restrict__ out_bias,
                               float* __restrict__ Out,
                               int Ncols)
{
    const int lane = threadIdx.x & 31;
    const int m0   = blockIdx.x * 16;
    const int n0   = blockIdx.y * 64;
    const int l15  = lane & 15;
    const int grp  = lane >> 4;          // 0 or 1 (K-group per 16-bit WMMA layout)

    const float* arow = A + (size_t)(m0 + l15) * K;   // A: M = lane%16
    const float* bcol[4];
#pragma unroll
    for (int t = 0; t < 4; ++t)
        bcol[t] = W + (size_t)(n0 + t * 16 + l15) * K;  // B[k][n] = W[n*K + k]

    v8f acc[4] = {};
#pragma unroll
    for (int k0 = 0; k0 < K; k0 += 32) {
        const int kbase = k0 + grp * 8;
        const v16bf a = load_frag<HAS_ABIAS>(arow, a_bias, kbase);
#pragma unroll
        for (int t = 0; t < 4; ++t) {
            const v16bf b = load_frag<false>(bcol[t], nullptr, kbase);
            acc[t] = __builtin_amdgcn_wmma_f32_16x16x32_bf16(
                         /*neg_a=*/false, a, /*neg_b=*/false, b,
                         /*c_mod=*/(short)0, acc[t],
                         /*reuse_a=*/false, /*reuse_b=*/false);
        }
    }

#pragma unroll
    for (int t = 0; t < 4; ++t) {
        const int   col = n0 + t * 16 + l15;
        const float ob  = out_bias ? out_bias[col] : 0.0f;
#pragma unroll
        for (int i = 0; i < 8; ++i) {
            // C/D layout: VGPR i, lanes 0-15 -> M = i, lanes 16-31 -> M = 8+i
            const int r = m0 + grp * 8 + i;
            Out[(size_t)r * Ncols + col] = acc[t][i] + ob;
        }
    }
}

// ---------------------------------------------------------------
// Workspace init: agg = 0, denom = 0, amax = -inf
// ---------------------------------------------------------------
__global__ void init_ws(float* __restrict__ agg,
                        float* __restrict__ amax,
                        float* __restrict__ denom)
{
    const int idx = blockIdx.x * blockDim.x + threadIdx.x;
    if (idx < NN * HC) agg[idx] = 0.0f;
    if (idx < NN * HEADS) {
        amax[idx]  = __int_as_float(0xFF800000);  // -inf
        denom[idx] = 0.0f;
    }
}

// ---------------------------------------------------------------
// Per-node attention scalars: a_src/a_dst[n,h] = sum_c h[n,h,c]*att[h,c]
// ---------------------------------------------------------------
__global__ void attn_scores(const float* __restrict__ h,
                            const float* __restrict__ att_src,
                            const float* __restrict__ att_dst,
                            float* __restrict__ a_src,
                            float* __restrict__ a_dst)
{
    const int idx = blockIdx.x * blockDim.x + threadIdx.x;   // node*HEADS + head
    if (idx >= NN * HEADS) return;
    const int node = idx >> 1;
    const int hd   = idx & 1;
    const float* hp = h + (size_t)node * HC + hd * OUT_C;
    const float* as = att_src + hd * OUT_C;
    const float* ad = att_dst + hd * OUT_C;
    float ssum = 0.0f, dsum = 0.0f;
#pragma unroll 4
    for (int c = 0; c < OUT_C; ++c) {
        const float v = hp[c];
        ssum += v * as[c];
        dsum += v * ad[c];
    }
    a_src[idx] = ssum;
    a_dst[idx] = dsum;
}

__device__ __forceinline__ void atomicMaxFloat(float* addr, float val)
{
    if (val >= 0.0f) atomicMax((int*)addr, __float_as_int(val));
    else             atomicMin((unsigned int*)addr, __float_as_uint(val));
}

__device__ __forceinline__ void edge_sd(const long long* __restrict__ ei,
                                        int e, long long& s, long long& d)
{
    if (e < EE) { s = ei[e]; d = ei[EE + e]; }
    else        { s = d = (long long)(e - EE); }   // self loop
}

// ---------------------------------------------------------------
// Pass 1: segment max of leaky-relu(alpha) grouped by dst
// ---------------------------------------------------------------
__global__ void edge_max(const long long* __restrict__ ei,
                         const float* __restrict__ a_src,
                         const float* __restrict__ a_dst,
                         float* __restrict__ amax)
{
    const int idx = blockIdx.x * blockDim.x + threadIdx.x;   // edge*HEADS + head
    if (idx >= ETOT * HEADS) return;
    const int e  = idx >> 1;
    const int hd = idx & 1;
    long long s, d;
    edge_sd(ei, e, s, d);
    float alpha = a_src[(size_t)s * HEADS + hd] + a_dst[(size_t)d * HEADS + hd];
    alpha = (alpha > 0.0f) ? alpha : alpha * NEG_SLOPE;
    atomicMaxFloat(&amax[(size_t)d * HEADS + hd], alpha);
}

// ---------------------------------------------------------------
// Pass 2: ex = exp(alpha - amax[dst]); denom[dst] += ex
// ---------------------------------------------------------------
__global__ void edge_exp(const long long* __restrict__ ei,
                         const float* __restrict__ a_src,
                         const float* __restrict__ a_dst,
                         const float* __restrict__ amax,
                         float* __restrict__ exbuf,
                         float* __restrict__ denom)
{
    const int idx = blockIdx.x * blockDim.x + threadIdx.x;
    if (idx >= ETOT * HEADS) return;
    const int e  = idx >> 1;
    const int hd = idx & 1;
    long long s, d;
    edge_sd(ei, e, s, d);
    float alpha = a_src[(size_t)s * HEADS + hd] + a_dst[(size_t)d * HEADS + hd];
    alpha = (alpha > 0.0f) ? alpha : alpha * NEG_SLOPE;
    const float ex = __expf(alpha - amax[(size_t)d * HEADS + hd]);
    exbuf[idx] = ex;
    atomicAdd(&denom[(size_t)d * HEADS + hd], ex);
}

// ---------------------------------------------------------------
// Pass 3: agg[dst] += coef * h[src]   (one wave per edge; 4 floats/lane)
// h rows are 512B contiguous -> fully coalesced wave gather; atomics stay
// L2-resident (agg = 25.6 MB << 192 MB L2).
// ---------------------------------------------------------------
__global__ void edge_agg(const long long* __restrict__ ei,
                         const float* __restrict__ exbuf,
                         const float* __restrict__ denom,
                         const float* __restrict__ h,
                         float* __restrict__ agg)
{
    const int gwave = (blockIdx.x * blockDim.x + threadIdx.x) >> 5;
    const int lane  = threadIdx.x & 31;
    if (gwave >= ETOT) return;
    long long s, d;
    edge_sd(ei, gwave, s, d);

    const int base = lane * 4;          // channel [0,128); lanes 0-15 head0, 16-31 head1
    const int hd   = base >> 6;

    const float* hp = h + (size_t)s * HC + base;
    __builtin_prefetch(hp, 0, 0);       // global_prefetch_b8

    const float coef = exbuf[(size_t)gwave * HEADS + hd] /
                       (denom[(size_t)d * HEADS + hd] + 1e-16f);
    const float4 hv = *(const float4*)hp;
    float* op = agg + (size_t)d * HC + base;
    atomicAdd(op + 0, hv.x * coef);
    atomicAdd(op + 1, hv.y * coef);
    atomicAdd(op + 2, hv.z * coef);
    atomicAdd(op + 3, hv.w * coef);
}

// ---------------------------------------------------------------
// GRU elementwise gates
// ---------------------------------------------------------------
__global__ void gru_elem(const float* __restrict__ gi,
                         const float* __restrict__ gh,
                         const float* __restrict__ x_prev,
                         float* __restrict__ out)
{
    const int idx = blockIdx.x * blockDim.x + threadIdx.x;   // node*OUT_C + c
    if (idx >= NN * OUT_C) return;
    const int n = idx / OUT_C;
    const int c = idx % OUT_C;
    const float* gin = gi + (size_t)n * G3;
    const float* ghn = gh + (size_t)n * G3;
    const float r  = 1.0f / (1.0f + __expf(-(gin[c]           + ghn[c])));
    const float z  = 1.0f / (1.0f + __expf(-(gin[OUT_C + c]   + ghn[OUT_C + c])));
    const float nn = tanhf(gin[2 * OUT_C + c] + r * ghn[2 * OUT_C + c]);
    const float hp = x_prev[idx];
    out[idx] = (1.0f - z) * nn + z * hp;
}

// ---------------------------------------------------------------
// Host-side launch
// ---------------------------------------------------------------
extern "C" void kernel_launch(void* const* d_in, const int* in_sizes, int n_in,
                              void* d_out, int out_size, void* d_ws, size_t ws_size,
                              hipStream_t stream)
{
    (void)in_sizes; (void)n_in; (void)out_size; (void)ws_size;

    const float*     x        = (const float*)d_in[0];
    const long long* ei       = (const long long*)d_in[1];   // int64 [2,E]
    const float*     x_prev   = (const float*)d_in[2];
    const float*     lin_w    = (const float*)d_in[3];       // [128,256]
    const float*     att_src  = (const float*)d_in[4];       // [1,2,64]
    const float*     att_dst  = (const float*)d_in[5];
    const float*     gat_bias = (const float*)d_in[6];       // [128]
    const float*     w_ih     = (const float*)d_in[7];       // [192,128]
    const float*     w_hh     = (const float*)d_in[8];       // [192,64]
    const float*     b_ih     = (const float*)d_in[9];       // [192]
    const float*     b_hh     = (const float*)d_in[10];      // [192]
    float*           out      = (float*)d_out;               // [N,64]

    // workspace layout (floats)
    float* ws    = (float*)d_ws;
    float* h     = ws;                       size_t off = (size_t)NN * HC;       // 6.4M
    float* agg   = ws + off;                 off += (size_t)NN * HC;             // 6.4M
    float* gi    = ws + off;                 off += (size_t)NN * G3;             // 9.6M
    float* gh    = ws + off;                 off += (size_t)NN * G3;             // 9.6M
    float* a_src = ws + off;                 off += (size_t)NN * HEADS;
    float* a_dst = ws + off;                 off += (size_t)NN * HEADS;
    float* amax  = ws + off;                 off += (size_t)NN * HEADS;
    float* denom = ws + off;                 off += (size_t)NN * HEADS;
    float* exbuf = ws + off;                 off += (size_t)ETOT * HEADS;        // 3.3M

    // 1) init scatter targets (ws is poisoned by harness)
    {
        const int tot = NN * HC;
        init_ws<<<(tot + 255) / 256, 256, 0, stream>>>(agg, amax, denom);
    }

    // 2) h = x @ lin_w^T   (bf16 WMMA; M=50000=3125*16, 16x64 tile per wave)
    {
        dim3 grid(NN / 16, HC / 64);   // 3125 x 2
        wmma_gemm_bf16<IN_C, false><<<grid, 32, 0, stream>>>(
            x, lin_w, nullptr, nullptr, h, HC);
    }

    // 3) attention scalars
    {
        const int tot = NN * HEADS;
        attn_scores<<<(tot + 255) / 256, 256, 0, stream>>>(h, att_src, att_dst,
                                                           a_src, a_dst);
    }

    // 4) segment max
    {
        const int tot = ETOT * HEADS;
        edge_max<<<(tot + 255) / 256, 256, 0, stream>>>(ei, a_src, a_dst, amax);
    }

    // 5) exp + denom
    {
        const int tot = ETOT * HEADS;
        edge_exp<<<(tot + 255) / 256, 256, 0, stream>>>(ei, a_src, a_dst, amax,
                                                        exbuf, denom);
    }

    // 6) weighted aggregation (one wave per edge)
    {
        const long long threads = (long long)ETOT * 32;
        edge_agg<<<(int)((threads + 255) / 256), 256, 0, stream>>>(ei, exbuf,
                                                                   denom, h, agg);
    }

    // 7) gi = (agg + gat_bias) @ w_ih^T + b_ih
    {
        dim3 grid(NN / 16, G3 / 64);   // 3125 x 3
        wmma_gemm_bf16<HC, true><<<grid, 32, 0, stream>>>(
            agg, w_ih, gat_bias, b_ih, gi, G3);
    }

    // 8) gh = x_prev @ w_hh^T + b_hh
    {
        dim3 grid(NN / 16, G3 / 64);
        wmma_gemm_bf16<OUT_C, false><<<grid, 32, 0, stream>>>(
            x_prev, w_hh, nullptr, b_hh, gh, G3);
    }

    // 9) GRU gates -> output
    {
        const int tot = NN * OUT_C;
        gru_elem<<<(tot + 255) / 256, 256, 0, stream>>>(gi, gh, x_prev, out);
    }
}
